// Encoder_65352222376280
// MI455X (gfx1250) — compile-verified, matrix operand order checked by reading, not run on hardware
//
#include <hip/hip_runtime.h>
#include <hip/hip_bf16.h>

// ---------------------------------------------------------------------------
// Problem constants (from the reference)
// ---------------------------------------------------------------------------
static constexpr int S    = 512;
static constexpr int Bt   = 8;
static constexpr int H    = 768;
static constexpr int H2   = 1536;   // 2*H
static constexpr int NH   = 12;
static constexpr int HD   = 64;
static constexpr int I    = 2048;
static constexpr int I2   = 4096;   // 2*I
static constexpr int L    = 6;
static constexpr int TOK  = S * Bt;       // 4096 tokens
static constexpr int BH   = Bt * NH;      // 96 attention batches
static constexpr int NBUK = 63;           // 2*BUCKET-1

typedef __attribute__((ext_vector_type(16))) __bf16 v16bf;
typedef __attribute__((ext_vector_type(8)))  float  v8f;
typedef __attribute__((ext_vector_type(4)))  int    v4i;

union FragBF { uint4 q[2]; v16bf v; };

__device__ __forceinline__ unsigned short f2bf(float f) {
    unsigned int u = __float_as_uint(f);
    unsigned int r = u + 0x7FFFu + ((u >> 16) & 1u);   // round-to-nearest-even
    return (unsigned short)(r >> 16);
}

// gfx1250 async global->LDS (ASYNCcnt-tracked, no VGPR staging) if available.
// Probe round 3 revealed the builtin signature: param0 = AS1 v4i* (non-const).
#if __has_builtin(__builtin_amdgcn_global_load_async_to_lds_b128) && \
    __has_builtin(__builtin_amdgcn_s_wait_asynccnt)
#define HAS_ASYNC_LDS 1
typedef __attribute__((address_space(1))) v4i* gp128;
typedef __attribute__((address_space(3))) v4i* lp128;
#else
#define HAS_ASYNC_LDS 0
#endif

// ---------------------------------------------------------------------------
// Tiled WMMA GEMM:  C[M,N] (+bias) = A[M,K](bf16) x B
//   BT = true : Bm is [N,K] row-major  -> computes A @ Bm^T  (weights, Q@K^T)
//   BT = false: Bm is [K,N] row-major  -> computes A @ Bm    (probs @ V)
//   GUARD     : M-edge guarding (only the M=63 pos projection needs it)
//
// Block: 256 threads (8 waves).  Block tile 256(M) x 64(N), K-step 32.
// Each wave owns a 32-row stripe: 2 A-frags x 4 B-frags = 8 v_wmma / K-step.
// Double-buffered LDS; tile k+1 is staged (async->LDS when available, else
// scalar-reg staged) while WMMAs run on tile k; one barrier per K-step.
// Requires: K % 32 == 0, N % 64 == 0 (true for every call below).
// ---------------------------------------------------------------------------
template <bool BT, bool GUARD>
__global__ __launch_bounds__(256, 1)
void gemm_wmma_bf16(const unsigned short* __restrict__ A,  long long strideA,
                    const unsigned short* __restrict__ Bm, long long strideB,
                    const float* __restrict__ bias,
                    float* __restrict__ C, long long strideC,
                    int M, int N, int K)
{
    __shared__ unsigned short As[2][256][40];   // [buf][m][k], +8 pad (80B rows)
    __shared__ unsigned short Bs[2][64][40];    // [buf][n][k]

    const int tid  = threadIdx.x;
    const int wave = tid >> 5;
    const int lane = tid & 31;
    const int half = lane >> 4;
    const int l16  = lane & 15;

    const int m0 = blockIdx.y * 256;
    const int n0 = blockIdx.x * 64;
    A  += (long long)blockIdx.z * strideA;
    Bm += (long long)blockIdx.z * strideB;
    C  += (long long)blockIdx.z * strideC;

    // staging coordinates (per thread)
    const int arow = tid >> 2;             // A/BT-B row; rows i are +64 apart
    const int acol = (tid & 3) * 8;        // 8-element octet within the 32-K tile
    const int bk   = tid >> 3;             // !BT: k index (0..31)
    const int bn   = (tid & 7) * 8;        // !BT: n octet
    const bool ok0 = !GUARD || (m0 + arow)       < M;
    const bool ok1 = !GUARD || (m0 + arow + 64)  < M;
    const bool ok2 = !GUARD || (m0 + arow + 128) < M;
    const bool ok3 = !GUARD || (m0 + arow + 192) < M;

    v8f acc[2][4];
    #pragma unroll
    for (int a = 0; a < 2; ++a)
        #pragma unroll
        for (int j = 0; j < 4; ++j)
            #pragma unroll
            for (int e = 0; e < 8; ++e) acc[a][j][e] = 0.0f;

    uint4 sb;                         // staged B octet (!BT, or sync path)
#if !HAS_ASYNC_LDS
    uint4 sa0, sa1, sa2, sa3;         // staged A octets (sync path, named scalars)
#endif

#if HAS_ASYNC_LDS
    auto stage_issue = [&](int k0, int buf) {
        const uint4 z{0u, 0u, 0u, 0u};
        if (ok0) __builtin_amdgcn_global_load_async_to_lds_b128(
                     (gp128)(A + (long long)(m0 + arow) * K + k0 + acol),
                     (lp128)&As[buf][arow][acol], 0, 0);
        else     *(uint4*)&As[buf][arow][acol] = z;
        if (ok1) __builtin_amdgcn_global_load_async_to_lds_b128(
                     (gp128)(A + (long long)(m0 + arow + 64) * K + k0 + acol),
                     (lp128)&As[buf][arow + 64][acol], 0, 0);
        else     *(uint4*)&As[buf][arow + 64][acol] = z;
        if (ok2) __builtin_amdgcn_global_load_async_to_lds_b128(
                     (gp128)(A + (long long)(m0 + arow + 128) * K + k0 + acol),
                     (lp128)&As[buf][arow + 128][acol], 0, 0);
        else     *(uint4*)&As[buf][arow + 128][acol] = z;
        if (ok3) __builtin_amdgcn_global_load_async_to_lds_b128(
                     (gp128)(A + (long long)(m0 + arow + 192) * K + k0 + acol),
                     (lp128)&As[buf][arow + 192][acol], 0, 0);
        else     *(uint4*)&As[buf][arow + 192][acol] = z;
        if constexpr (BT) {
            __builtin_amdgcn_global_load_async_to_lds_b128(
                (gp128)(Bm + (long long)(n0 + arow) * K + k0 + acol),
                (lp128)&Bs[buf][arow][acol], 0, 0);
        } else {
            sb = *(const uint4*)(Bm + (long long)(k0 + bk) * N + n0 + bn);
        }
    };
    auto stage_commit = [&](int buf) {
        if constexpr (!BT) {   // transpose-scatter the B octet
            const unsigned short* e = (const unsigned short*)&sb;
            #pragma unroll
            for (int t = 0; t < 8; ++t) Bs[buf][bn + t][bk] = e[t];
        }
        __builtin_amdgcn_s_wait_asynccnt(0);
    };
#else
    auto stage_issue = [&](int k0, int /*buf*/) {
        const uint4 z{0u, 0u, 0u, 0u};
        sa0 = ok0 ? *(const uint4*)(A + (long long)(m0 + arow) * K + k0 + acol)       : z;
        sa1 = ok1 ? *(const uint4*)(A + (long long)(m0 + arow + 64) * K + k0 + acol)  : z;
        sa2 = ok2 ? *(const uint4*)(A + (long long)(m0 + arow + 128) * K + k0 + acol) : z;
        sa3 = ok3 ? *(const uint4*)(A + (long long)(m0 + arow + 192) * K + k0 + acol) : z;
        if constexpr (BT)
            sb = *(const uint4*)(Bm + (long long)(n0 + arow) * K + k0 + acol);
        else
            sb = *(const uint4*)(Bm + (long long)(k0 + bk) * N + n0 + bn);
    };
    auto stage_commit = [&](int buf) {
        *(uint4*)&As[buf][arow][acol]       = sa0;
        *(uint4*)&As[buf][arow + 64][acol]  = sa1;
        *(uint4*)&As[buf][arow + 128][acol] = sa2;
        *(uint4*)&As[buf][arow + 192][acol] = sa3;
        if constexpr (BT) {
            *(uint4*)&Bs[buf][arow][acol] = sb;
        } else {
            const unsigned short* e = (const unsigned short*)&sb;
            #pragma unroll
            for (int t = 0; t < 8; ++t) Bs[buf][bn + t][bk] = e[t];
        }
    };
#endif

    const int nk = K >> 5;
    stage_issue(0, 0);
    stage_commit(0);
    __syncthreads();

    for (int kt = 0; kt < nk; ++kt) {
        const int buf = kt & 1;

        if (kt + 1 < nk) {
            stage_issue((kt + 1) << 5, buf ^ 1);  // overlaps WMMAs below
            if (kt + 2 < nk) {                    // global_prefetch_b8, 2 tiles ahead
                int k2 = (kt + 2) << 5;
                if (ok0) __builtin_prefetch(A + (long long)(m0 + arow) * K + k2, 0, 0);
                if constexpr (BT)
                    __builtin_prefetch(Bm + (long long)(n0 + arow) * K + k2, 0, 0);
                else
                    __builtin_prefetch(Bm + (long long)(k2 + bk) * N + n0, 0, 0);
            }
        }

        // ---- fragments (16-bit A-matrix lane layout, ISA 7.12.2):
        // lane l: m/n = l&15; elems 0..7 -> K = 8*half+e, elems 8..15 -> 16+8*half+(e-8)
        FragBF fa[2], fb[4];
        const int mw = wave * 32;
        #pragma unroll
        for (int a = 0; a < 2; ++a) {
            fa[a].q[0] = *(const uint4*)&As[buf][mw + a * 16 + l16][half * 8];
            fa[a].q[1] = *(const uint4*)&As[buf][mw + a * 16 + l16][16 + half * 8];
        }
        #pragma unroll
        for (int j = 0; j < 4; ++j) {
            fb[j].q[0] = *(const uint4*)&Bs[buf][j * 16 + l16][half * 8];
            fb[j].q[1] = *(const uint4*)&Bs[buf][j * 16 + l16][16 + half * 8];
        }
        #pragma unroll
        for (int a = 0; a < 2; ++a)
            #pragma unroll
            for (int j = 0; j < 4; ++j)
                acc[a][j] = __builtin_amdgcn_wmma_f32_16x16x32_bf16(
                    false, fa[a].v, false, fb[j].v, (short)0, acc[a][j], false, false);

        if (kt + 1 < nk) stage_commit(buf ^ 1);
        __syncthreads();
    }

    // ---- store (C/D layout: lane n = l&15, VGPR v -> row v + 8*half) ----
    #pragma unroll
    for (int a = 0; a < 2; ++a) {
        #pragma unroll
        for (int j = 0; j < 4; ++j) {
            int col = n0 + j * 16 + l16;
            float bv = bias ? bias[col] : 0.0f;
            #pragma unroll
            for (int v = 0; v < 8; ++v) {
                int row = m0 + wave * 32 + a * 16 + half * 8 + v;
                if (!GUARD || row < M)
                    C[(long long)row * N + col] = acc[a][j][v] + bv;
            }
        }
    }
}

// ---------------------------------------------------------------------------
// LayerNorm (optionally affine) -> bf16.  One block per row.
// ---------------------------------------------------------------------------
__global__ __launch_bounds__(256)
void k_layernorm_bf16(const float* __restrict__ x, const float* __restrict__ g,
                      const float* __restrict__ bta, unsigned short* __restrict__ out,
                      int D)
{
    __shared__ float s1[256], s2[256];
    long long row = blockIdx.x;
    const float* xr = x + row * D;
    float s = 0.f, ss = 0.f;
    for (int c = threadIdx.x; c < D; c += 256) { float v = xr[c]; s += v; ss += v * v; }
    s1[threadIdx.x] = s; s2[threadIdx.x] = ss;
    __syncthreads();
    for (int off = 128; off > 0; off >>= 1) {
        if ((int)threadIdx.x < off) {
            s1[threadIdx.x] += s1[threadIdx.x + off];
            s2[threadIdx.x] += s2[threadIdx.x + off];
        }
        __syncthreads();
    }
    float mu  = s1[0] / D;
    float var = s2[0] / D - mu * mu;
    float rs  = rsqrtf(var + 1e-7f);
    unsigned short* orow = out + row * D;
    for (int c = threadIdx.x; c < D; c += 256) {
        float y = (xr[c] - mu) * rs;
        if (g) y = y * g[c] + bta[c];
        orow[c] = f2bf(y);
    }
}

// x += LN(d) * g + b   (one block per row)
__global__ __launch_bounds__(256)
void k_residual_ln(float* __restrict__ x, const float* __restrict__ d,
                   const float* __restrict__ g, const float* __restrict__ bta, int D)
{
    __shared__ float s1[256], s2[256];
    long long row = blockIdx.x;
    const float* dr = d + row * D;
    float s = 0.f, ss = 0.f;
    for (int c = threadIdx.x; c < D; c += 256) { float v = dr[c]; s += v; ss += v * v; }
    s1[threadIdx.x] = s; s2[threadIdx.x] = ss;
    __syncthreads();
    for (int off = 128; off > 0; off >>= 1) {
        if ((int)threadIdx.x < off) {
            s1[threadIdx.x] += s1[threadIdx.x + off];
            s2[threadIdx.x] += s2[threadIdx.x + off];
        }
        __syncthreads();
    }
    float mu  = s1[0] / D;
    float var = s2[0] / D - mu * mu;
    float rs  = rsqrtf(var + 1e-7f);
    float* xr = x + row * D;
    for (int c = threadIdx.x; c < D; c += 256)
        xr[c] += (dr[c] - mu) * rs * g[c] + bta[c];
}

// ---------------------------------------------------------------------------
// Positional bucket projections:
//   C1[bh,q,j] = sum_d q[q,b,h,d]  * kpos[j,h,d]
//   C2[bh,k,j] = sum_d k[k,b,h,d]  * qpos[j,h,d]
// ---------------------------------------------------------------------------
__global__ __launch_bounds__(64)
void k_c1c2(const float* __restrict__ qk, const float* __restrict__ pos,
            float* __restrict__ C1, float* __restrict__ C2)
{
    int blk = blockIdx.x;          // bh*S + row
    int row = blk & (S - 1);
    int bh  = blk >> 9;
    int b = bh / NH, h = bh - b * NH;
    int j = threadIdx.x;
    const float* qrow = qk + ((long long)row * Bt + b) * H2 + h * HD;  // q part
    const float* krow = qrow + H;                                      // k part
    float a1 = 0.f, a2 = 0.f;
    if (j < NBUK) {
        const float* pj = pos + (long long)j * H2 + h * 128;
        #pragma unroll 8
        for (int d = 0; d < HD; ++d) {
            a1 += qrow[d] * pj[HD + d];   // kpos
            a2 += krow[d] * pj[d];        // qpos
        }
    }
    C1[(long long)blk * 64 + j] = a1;
    C2[(long long)blk * 64 + j] = a2;
}

// ---------------------------------------------------------------------------
// Softmax over keys with positional gather + mask; writes bf16 probs.
// ---------------------------------------------------------------------------
__global__ __launch_bounds__(256)
void k_softmax(const float* __restrict__ SC, const float* __restrict__ C1,
               const float* __restrict__ C2, const int* __restrict__ idx,
               const unsigned char* __restrict__ mask,
               unsigned short* __restrict__ PB, float scale)
{
    __shared__ float red[256];
    int rowid = blockIdx.x;            // bh*S + q
    int q  = rowid & (S - 1);
    int bh = rowid >> 9;
    int b  = bh / NH;
    const float* sc = SC + (long long)rowid * S;
    const float* c1 = C1 + (long long)rowid * 64;
    const float* c2 = C2 + (long long)bh * S * 64;
    const int* ir = idx + (long long)q * S;
    const unsigned char* mr = mask + (long long)b * S;

    float vals[2];
    #pragma unroll
    for (int i = 0; i < 2; ++i) {
        int k = threadIdx.x + i * 256;
        int j = ir[k];
        float v = (sc[k] + c1[j] + c2[(long long)k * 64 + j]) * scale;
        if (mr[k]) v = -INFINITY;
        vals[i] = v;
    }
    float m = fmaxf(vals[0], vals[1]);
    red[threadIdx.x] = m; __syncthreads();
    for (int off = 128; off > 0; off >>= 1) {
        if ((int)threadIdx.x < off)
            red[threadIdx.x] = fmaxf(red[threadIdx.x], red[threadIdx.x + off]);
        __syncthreads();
    }
    float mx = red[0];
    __syncthreads();
    float e0 = __expf(vals[0] - mx), e1 = __expf(vals[1] - mx);
    red[threadIdx.x] = e0 + e1; __syncthreads();
    for (int off = 128; off > 0; off >>= 1) {
        if ((int)threadIdx.x < off)
            red[threadIdx.x] += red[threadIdx.x + off];
        __syncthreads();
    }
    float inv = 1.0f / red[0];
    unsigned short* pr = PB + (long long)rowid * S;
    pr[threadIdx.x]       = f2bf(mr[threadIdx.x]       ? 0.f : e0 * inv);
    pr[threadIdx.x + 256] = f2bf(mr[threadIdx.x + 256] ? 0.f : e1 * inv);
}

// ---------------------------------------------------------------------------
// Layout shuffles + misc elementwise
// ---------------------------------------------------------------------------
__global__ __launch_bounds__(256)
void k_pack_qkv(const float* __restrict__ qk, const float* __restrict__ v,
                unsigned short* __restrict__ Qb, unsigned short* __restrict__ Kb,
                unsigned short* __restrict__ Vb)
{
    long long tid = (long long)blockIdx.x * 256 + threadIdx.x;
    if (tid >= (long long)TOK * H) return;
    int c = (int)(tid % H);
    long long t = tid / H;
    int b = (int)(t % Bt);
    int s = (int)(t / Bt);
    int h = c >> 6, d = c & 63;
    long long o = (((long long)(b * NH + h)) * S + s) * HD + d;
    Qb[o] = f2bf(qk[t * H2 + c]);
    Kb[o] = f2bf(qk[t * H2 + H + c]);
    Vb[o] = f2bf(v[t * H + c]);
}

__global__ __launch_bounds__(256)
void k_unpack_ctx(const float* __restrict__ ctx, unsigned short* __restrict__ out)
{
    long long tid = (long long)blockIdx.x * 256 + threadIdx.x;
    if (tid >= (long long)TOK * H) return;
    int c = (int)(tid % H);
    long long t = tid / H;
    int b = (int)(t % Bt);
    int s = (int)(t / Bt);
    int h = c >> 6, d = c & 63;
    out[tid] = f2bf(ctx[(((long long)(b * NH + h)) * S + s) * HD + d]);
}

__global__ __launch_bounds__(256)
void k_geglu(const float* __restrict__ u, float* __restrict__ out)
{
    long long tid = (long long)blockIdx.x * 256 + threadIdx.x;
    if (tid >= (long long)TOK * I) return;
    long long t = tid / I;
    int i = (int)(tid % I);
    float a  = u[t * (long long)I2 + i];
    float gt = u[t * (long long)I2 + I + i];
    float tn = tanhf(0.7978845608028654f * (gt + 0.044715f * gt * gt * gt));
    out[tid] = a * 0.5f * gt * (1.0f + tn);
}

__global__ __launch_bounds__(256)
void k_f2bf(const float* __restrict__ in, unsigned short* __restrict__ out, long long n)
{
    long long i = (long long)blockIdx.x * 256 + threadIdx.x;
    long long st = (long long)gridDim.x * 256;
    for (; i < n; i += st) out[i] = f2bf(in[i]);
}

__global__ __launch_bounds__(256)
void k_addf(float* __restrict__ x, const float* __restrict__ d, long long n)
{
    long long i = (long long)blockIdx.x * 256 + threadIdx.x;
    long long st = (long long)gridDim.x * 256;
    for (; i < n; i += st) x[i] += d[i];
}

// ---------------------------------------------------------------------------
// Host-side orchestration
// ---------------------------------------------------------------------------
static inline long long cdivll(long long a, long long b) { return (a + b - 1) / b; }

extern "C" void kernel_launch(void* const* d_in, const int* in_sizes, int n_in,
                              void* d_out, int out_size, void* d_ws, size_t ws_size,
                              hipStream_t stream)
{
    (void)in_sizes; (void)n_in; (void)out_size;

    const float* hs   = (const float*)d_in[0];
    const unsigned char* amask = (const unsigned char*)d_in[1];
    const float* rel  = (const float*)d_in[2];
    const float* Wqk  = (const float*)d_in[3];
    const float* bqk  = (const float*)d_in[4];
    const float* Wv   = (const float*)d_in[5];
    const float* bv   = (const float*)d_in[6];
    const float* Wo   = (const float*)d_in[7];
    const float* bo   = (const float*)d_in[8];
    const float* lng  = (const float*)d_in[9];
    const float* lnb  = (const float*)d_in[10];
    const float* W1   = (const float*)d_in[11];
    const float* W2   = (const float*)d_in[12];
    const int*   pidx = (const int*)d_in[13];

    // ---- workspace layout (all offsets multiples of 256B) ----
    char* w = (char*)d_ws;
    size_t off = 0;
    auto take = [&](size_t bytes) { char* p = w + off; off += (bytes + 255) & ~(size_t)255; return p; };

    float*          X    = (float*)         take((size_t)TOK * H  * 4);   // residual stream
    unsigned short* HB   = (unsigned short*)take((size_t)TOK * I  * 2);   // LN outputs (bf16)
    float*          QK   = (float*)         take((size_t)TOK * H2 * 4);
    float*          V    = (float*)         take((size_t)TOK * H  * 4);
    float*          POS  = (float*)         take((size_t)NBUK * H2 * 4);
    unsigned short* REB  = (unsigned short*)take((size_t)NBUK * H  * 2);
    unsigned short* Qb   = (unsigned short*)take((size_t)BH * S * HD * 2);
    unsigned short* Kb   = (unsigned short*)take((size_t)BH * S * HD * 2);
    unsigned short* Vb   = (unsigned short*)take((size_t)BH * S * HD * 2);
    float*          SC   = (float*)         take((size_t)BH * S * S * 4); // aliased by U later
    float*          C1   = (float*)         take((size_t)BH * S * 64 * 4);
    float*          C2   = (float*)         take((size_t)BH * S * 64 * 4);
    unsigned short* PB   = (unsigned short*)take((size_t)BH * S * S * 2); // aliased by UA later
    float*          CTX  = (float*)         take((size_t)BH * S * HD * 4);
    unsigned short* CTXB = (unsigned short*)take((size_t)TOK * H * 2);
    float*          D1   = (float*)         take((size_t)TOK * H * 4);
    unsigned short* WL   = (unsigned short*)take((size_t)(2*H*H + H*H + H*H + 2*I*H + H*I) * 2);

    // aliases (lifetimes disjoint within a layer)
    float* U  = SC;          // TOK * I2 fp32  (67MB <= 100MB)
    float* UA = (float*)PB;  // TOK * I  fp32  (33.5MB <= 50MB)

    if (ws_size < off) {
        // Not enough scratch: degrade gracefully (identity) rather than fault.
        (void)hipMemcpyAsync(d_out, hs, (size_t)TOK * H * 4, hipMemcpyDeviceToDevice, stream);
        return;
    }

    unsigned short* WLqk = WL;
    unsigned short* WLv  = WLqk + (size_t)2 * H * H;
    unsigned short* WLo  = WLv  + (size_t)H * H;
    unsigned short* WL1  = WLo  + (size_t)H * H;
    unsigned short* WL2  = WL1  + (size_t)2 * I * H;

    const float scale = 1.0f / sqrtf(3.0f * HD);

    // X = hidden_states ; REB = bf16(relative_embedding)
    (void)hipMemcpyAsync(X, hs, (size_t)TOK * H * 4, hipMemcpyDeviceToDevice, stream);
    k_f2bf<<<dim3(256), dim3(256), 0, stream>>>(rel, REB, (long long)NBUK * H);

    for (int l = 0; l < L; ++l) {
        const float* Wqk_l = Wqk + (size_t)l * 2 * H * H;
        const float* bqk_l = bqk + (size_t)l * 2 * H;
        const float* Wv_l  = Wv  + (size_t)l * H * H;
        const float* bv_l  = bv  + (size_t)l * H;
        const float* Wo_l  = Wo  + (size_t)l * H * H;
        const float* bo_l  = bo  + (size_t)l * H;
        const float* g_l   = lng + (size_t)l * H;
        const float* b_l   = lnb + (size_t)l * H;
        const float* W1_l  = W1  + (size_t)l * 2 * I * H;
        const float* W2_l  = W2  + (size_t)l * H * I;

        // convert this layer's weights to bf16
        k_f2bf<<<dim3(2048), dim3(256), 0, stream>>>(Wqk_l, WLqk, (long long)2 * H * H);
        k_f2bf<<<dim3(2048), dim3(256), 0, stream>>>(Wv_l,  WLv,  (long long)H * H);
        k_f2bf<<<dim3(2048), dim3(256), 0, stream>>>(Wo_l,  WLo,  (long long)H * H);
        k_f2bf<<<dim3(4096), dim3(256), 0, stream>>>(W1_l,  WL1,  (long long)2 * I * H);
        k_f2bf<<<dim3(4096), dim3(256), 0, stream>>>(W2_l,  WL2,  (long long)H * I);

        // h = LN(x) -> bf16
        k_layernorm_bf16<<<dim3(TOK), dim3(256), 0, stream>>>(X, nullptr, nullptr, HB, H);

        // qk = h @ Wqk^T + bqk ; v = h @ Wv^T + bv
        gemm_wmma_bf16<true, false><<<dim3(H2/64, TOK/256, 1), dim3(256), 0, stream>>>(
            HB, 0, WLqk, 0, bqk_l, QK, 0, TOK, H2, H);
        gemm_wmma_bf16<true, false><<<dim3(H/64, TOK/256, 1), dim3(256), 0, stream>>>(
            HB, 0, WLv, 0, bv_l, V, 0, TOK, H, H);

        // pos = relative_embedding @ Wqk^T + bqk   (M=63, guarded)
        gemm_wmma_bf16<true, true><<<dim3(H2/64, 1, 1), dim3(256), 0, stream>>>(
            REB, 0, WLqk, 0, bqk_l, POS, 0, NBUK, H2, H);

        // pack heads: [B,NH,S,hd] bf16
        k_pack_qkv<<<dim3(cdivll((long long)TOK * H, 256)), dim3(256), 0, stream>>>(QK, V, Qb, Kb, Vb);

        // content scores: Q @ K^T per (b,h)
        gemm_wmma_bf16<true, false><<<dim3(S/64, S/256, BH), dim3(256), 0, stream>>>(
            Qb, (long long)S * HD, Kb, (long long)S * HD, nullptr,
            SC, (long long)S * S, S, S, HD);

        // positional bucket dot products
        k_c1c2<<<dim3(BH * S), dim3(64), 0, stream>>>(QK, POS, C1, C2);

        // softmax with positional gather & mask -> bf16 probs
        k_softmax<<<dim3(BH * S), dim3(256), 0, stream>>>(SC, C1, C2, pidx, amask, PB, scale);

        // ctx = probs @ V
        gemm_wmma_bf16<false, false><<<dim3(HD/64, S/256, BH), dim3(256), 0, stream>>>(
            PB, (long long)S * S, Vb, (long long)S * HD, nullptr,
            CTX, (long long)S * HD, S, HD, S);

        // back to [S,B,H] bf16, then output projection
        k_unpack_ctx<<<dim3(cdivll((long long)TOK * H, 256)), dim3(256), 0, stream>>>(CTX, CTXB);
        gemm_wmma_bf16<true, false><<<dim3(H/64, TOK/256, 1), dim3(256), 0, stream>>>(
            CTXB, 0, WLo, 0, bo_l, D1, 0, TOK, H, H);

        // x += LN(ctx)*g + b
        k_residual_ln<<<dim3(TOK), dim3(256), 0, stream>>>(X, D1, g_l, b_l, H);

        // ---- FFN: u = LN(x) @ W1^T ; a*gelu(gate) ; LN ; @ W2^T ; residual ----
        k_layernorm_bf16<<<dim3(TOK), dim3(256), 0, stream>>>(X, nullptr, nullptr, HB, H);
        gemm_wmma_bf16<true, false><<<dim3(I2/64, TOK/256, 1), dim3(256), 0, stream>>>(
            HB, 0, WL1, 0, nullptr, U, 0, TOK, I2, H);
        k_geglu<<<dim3(cdivll((long long)TOK * I, 256)), dim3(256), 0, stream>>>(U, UA);
        k_layernorm_bf16<<<dim3(TOK), dim3(256), 0, stream>>>(UA, nullptr, nullptr, HB, I);
        gemm_wmma_bf16<true, false><<<dim3(H/64, TOK/256, 1), dim3(256), 0, stream>>>(
            HB, 0, WL2, 0, nullptr, D1, 0, TOK, H, I);
        k_addf<<<dim3(4096), dim3(256), 0, stream>>>(X, D1, (long long)TOK * H);
    }

    (void)hipMemcpyAsync(d_out, X, (size_t)TOK * H * 4, hipMemcpyDeviceToDevice, stream);
}
